// LinearAttention_43954695307849
// MI455X (gfx1250) — matrix operand order repaired and checked by reading, not compiled, not run on hardware
//
#include <hip/hip_runtime.h>
#include <hip/hip_bf16.h>

// ---------------------------------------------------------------------------
// Causal linear attention (ELU+1 feature map) for MI455X / gfx1250.
// Pipeline: [qkv GEMM + feature map] -> [chunked causal linear scan] -> [out GEMM]
// Matrix math: v_wmma_f32_16x16x32_bf16.  Tile staging: TDM tensor_load_to_lds
// (TENSORcnt-tracked DMA with hardware LDS row padding).
// ---------------------------------------------------------------------------

typedef __attribute__((ext_vector_type(16))) __bf16 v16bf;
typedef __attribute__((ext_vector_type(8)))  float  v8f;
typedef unsigned int u32x4 __attribute__((ext_vector_type(4)));
typedef int          i32x8 __attribute__((ext_vector_type(8)));
typedef int          i32x4 __attribute__((ext_vector_type(4)));

constexpr int Bc  = 4;
constexpr int Tc  = 4096;
constexpr int Dc  = 1024;   // model dim = H*HD
constexpr int Hc  = 16;
constexpr int HDc = 64;
constexpr int Nc  = Bc * Tc;          // 16384 rows
constexpr float EPSc = 1e-6f;

__device__ __forceinline__ v8f wmma_bf(v16bf a, v16bf b, v8f c) {
  return __builtin_amdgcn_wmma_f32_16x16x32_bf16(
      false, a, false, b, (short)0, c, false, false);
}

__device__ __forceinline__ v8f vzero() {
  v8f z;
#pragma unroll
  for (int i = 0; i < 8; ++i) z[i] = 0.f;
  return z;
}

__device__ __forceinline__ float feature_map(float x) {
  return x > 0.f ? (x + 1.f) : __expf(x);   // elu(x)+1
}

// ---------------------------------------------------------------------------
// TDM: 2D tile DMA global -> LDS.  Descriptor per CDNA5 ISA.
//   data_size_code: 0=1B,1=2B,2=4B,3=8B
//   pad_interval_code: pad after 2^(v+1) DWORDs; pad_amount_code: pad (v+1) DWORDs
// Tensor dims == tile dims (tiles are fully in-bounds by construction).
// This toolchain exposes the 6-arg builtin: (g0, g1, g2, g3, g_extra, cpol).
// ---------------------------------------------------------------------------
__device__ __forceinline__ void tdm_load_2d(unsigned lds_addr, const void* gptr,
                                            unsigned data_size_code,
                                            unsigned tile_d0, unsigned tile_d1,
                                            unsigned long long row_stride_elems,
                                            unsigned pad_interval_code,
                                            unsigned pad_amount_code) {
  const unsigned long long ga = (unsigned long long)(uintptr_t)gptr;
  u32x4 g0;
  g0[0] = 1u;                                              // count=1, no gather
  g0[1] = lds_addr;                                        // LDS byte address
  g0[2] = (unsigned)(ga & 0xFFFFFFFFu);                    // global_addr[31:0]
  g0[3] = (unsigned)((ga >> 32) & 0x1FFFFFFu) | 0x80000000u; // [56:32] | type=2
  i32x8 g1;
  g1[0] = (int)((data_size_code << 16) | (1u << 20) |      // pad_enable
                (pad_interval_code << 22) | (pad_amount_code << 25));
  g1[1] = (int)((tile_d0 & 0xFFFFu) << 16);                // tensor_dim0[15:0]
  g1[2] = (int)((tile_d0 >> 16) | ((tile_d1 & 0xFFFFu) << 16)); // td0 hi | td1 lo
  g1[3] = (int)((tile_d1 >> 16) | (tile_d0 << 16));        // td1 hi | tile_dim0
  g1[4] = (int)(tile_d1 & 0xFFFFu);                        // tile_dim1 (tile_dim2=0)
  g1[5] = (int)(row_stride_elems & 0xFFFFFFFFull);         // dim0_stride[31:0]
  g1[6] = (int)((row_stride_elems >> 32) & 0xFFFFull);     // dim0_stride[47:32]
  g1[7] = 0;
  i32x4 gz4;
  gz4[0] = 0; gz4[1] = 0; gz4[2] = 0; gz4[3] = 0;
  i32x8 gz8;
#pragma unroll
  for (int i = 0; i < 8; ++i) gz8[i] = 0;
  __builtin_amdgcn_tensor_load_to_lds(g0, g1, gz4, gz4, gz8, 0);
}

// ---------------- fragment builders (CDNA5 wave32 WMMA layouts) -------------
// A 16x32 bf16 from row-major bf16 p[row*ld + k]
__device__ __forceinline__ v16bf afrag(const __bf16* p, int ld, int row, int k0, int lane) {
  v16bf a; const int kb = (lane >> 4) * 8;
#pragma unroll
  for (int v = 0; v < 8; ++v) {
    const int k = k0 + ((v & 4) ? 16 : 0) + kb + 2 * (v & 3);
    a[2 * v]     = p[row * ld + k];
    a[2 * v + 1] = p[row * ld + k + 1];
  }
  return a;
}
// A 16x32 bf16 from transposed bf16 p[k*ld + row]
__device__ __forceinline__ v16bf afragT(const __bf16* p, int ld, int row, int k0, int lane) {
  v16bf a; const int kb = (lane >> 4) * 8;
#pragma unroll
  for (int v = 0; v < 8; ++v) {
    const int k = k0 + ((v & 4) ? 16 : 0) + kb + 2 * (v & 3);
    a[2 * v]     = p[k * ld + row];
    a[2 * v + 1] = p[(k + 1) * ld + row];
  }
  return a;
}
// A 16x32 from row-major f32 source (convert)
__device__ __forceinline__ v16bf afragF(const float* p, int ld, int row, int k0, int lane) {
  v16bf a; const int kb = (lane >> 4) * 8;
#pragma unroll
  for (int v = 0; v < 8; ++v) {
    const int k = k0 + ((v & 4) ? 16 : 0) + kb + 2 * (v & 3);
    a[2 * v]     = (__bf16)p[row * ld + k];
    a[2 * v + 1] = (__bf16)p[row * ld + k + 1];
  }
  return a;
}
// B 32x16 bf16 from K-contiguous bf16 p[col*ld + k]
__device__ __forceinline__ v16bf bfragT(const __bf16* p, int ld, int col, int k0, int lane) {
  v16bf b; const int kb = (lane >> 4) * 8;
#pragma unroll
  for (int v = 0; v < 8; ++v) {
    const int k = k0 + ((v & 4) ? 16 : 0) + kb + 2 * (v & 3);
    b[2 * v]     = p[col * ld + k];
    b[2 * v + 1] = p[col * ld + k + 1];
  }
  return b;
}
// B 32x16 bf16 from row-major-in-K bf16 p[k*ld + col]
__device__ __forceinline__ v16bf bfragR(const __bf16* p, int ld, int col, int k0, int lane) {
  v16bf b; const int kb = (lane >> 4) * 8;
#pragma unroll
  for (int v = 0; v < 8; ++v) {
    const int k = k0 + ((v & 4) ? 16 : 0) + kb + 2 * (v & 3);
    b[2 * v]     = p[k * ld + col];
    b[2 * v + 1] = p[(k + 1) * ld + col];
  }
  return b;
}
// B 32x16 from row-major-in-K f32 p[k*ld + col] (convert)
__device__ __forceinline__ v16bf bfragRF(const float* p, int ld, int col, int k0, int lane) {
  v16bf b; const int kb = (lane >> 4) * 8;
#pragma unroll
  for (int v = 0; v < 8; ++v) {
    const int k = k0 + ((v & 4) ? 16 : 0) + kb + 2 * (v & 3);
    b[2 * v]     = (__bf16)p[k * ld + col];
    b[2 * v + 1] = (__bf16)p[(k + 1) * ld + col];
  }
  return b;
}

// ---------------------------------------------------------------------------
// Kernel 1: Q/K/V projection.  grid = (128 row-tiles, 8 col-tiles, 3 matrices)
// x tile (128x32 f32) and W tile (32x128 f32) DMA'd into LDS by the TDM.
// ---------------------------------------------------------------------------
__global__ __launch_bounds__(256)
void qkv_proj_kernel(const float* __restrict__ x,
                     const float* __restrict__ Wq,
                     const float* __restrict__ Wk,
                     const float* __restrict__ Wv,
                     __bf16* __restrict__ Q,
                     __bf16* __restrict__ K,
                     __bf16* __restrict__ V) {
  constexpr int LDX = 34;   // f32 stride: 32 DW rows + 2 DW TDM pad (conflict-free)
  constexpr int LDW = 130;  // f32 stride: 128 DW rows + 2 DW TDM pad
  __shared__ float sX[128 * LDX];   // [row][k]
  __shared__ float sW[32 * LDW];    // [k][col]

  const int rowTile = blockIdx.x;
  const int colTile = blockIdx.y;
  const int mat     = blockIdx.z;          // 0:Q 1:K 2:V
  const float*  W   = (mat == 0) ? Wq : (mat == 1) ? Wk : Wv;
  __bf16*       Out = (mat == 0) ? Q  : (mat == 1) ? K  : V;

  const int tid  = threadIdx.x;
  const int lane = tid & 31;
  const int wave = tid >> 5;
  const int wm   = wave >> 2;              // 0..1 -> 64-row strip
  const int wn   = wave & 3;               // 0..3 -> 32-col strip

  v8f acc[4][2];
#pragma unroll
  for (int mt = 0; mt < 4; ++mt)
#pragma unroll
    for (int nt = 0; nt < 2; ++nt) acc[mt][nt] = vzero();

  for (int k0 = 0; k0 < Dc; k0 += 32) {
    if (wave == 0) {
      // x tile: rows rowTile*128.., cols k0..k0+31 (row = 32 DW -> interval code 4)
      tdm_load_2d((unsigned)(uintptr_t)sX,
                  x + (size_t)(rowTile * 128) * Dc + k0,
                  /*4B*/2, /*d0*/32, /*d1*/128, /*stride*/Dc, /*intv*/4, /*pad*/1);
      // W tile: rows k0..k0+31, cols colTile*128.. (row = 128 DW -> code 6)
      tdm_load_2d((unsigned)(uintptr_t)sW,
                  W + (size_t)k0 * Dc + colTile * 128,
                  /*4B*/2, /*d0*/128, /*d1*/32, /*stride*/Dc, /*intv*/6, /*pad*/1);
      __builtin_amdgcn_s_wait_tensorcnt(0);
    }
    __syncthreads();

    v16bf bfr[2];
#pragma unroll
    for (int nt = 0; nt < 2; ++nt)
      bfr[nt] = bfragRF(sW, LDW, wn * 32 + nt * 16 + (lane & 15), 0, lane);
#pragma unroll
    for (int mt = 0; mt < 4; ++mt) {
      const v16bf afr = afragF(sX, LDX, wm * 64 + mt * 16 + (lane & 15), 0, lane);
#pragma unroll
      for (int nt = 0; nt < 2; ++nt) acc[mt][nt] = wmma_bf(afr, bfr[nt], acc[mt][nt]);
    }
    __syncthreads();
  }

  const int mhalf = (lane >> 4) * 8;
  const int n     = lane & 15;
#pragma unroll
  for (int mt = 0; mt < 4; ++mt)
#pragma unroll
    for (int nt = 0; nt < 2; ++nt)
#pragma unroll
      for (int i = 0; i < 8; ++i) {
        const int grow = rowTile * 128 + wm * 64 + mt * 16 + i + mhalf;  // b*T + t
        const int gcol = colTile * 128 + wn * 32 + nt * 16 + n;          // h*64 + hd
        float val = acc[mt][nt][i];
        if (mat != 2) val = feature_map(val);
        const int b = grow >> 12, t = grow & (Tc - 1);
        const int h = gcol >> 6,  hd = gcol & (HDc - 1);
        Out[(((size_t)(b * Hc + h) * Tc + t) << 6) + hd] = (__bf16)val;
      }
}

// ---------------------------------------------------------------------------
// Kernel 2: chunked causal linear-attention scan. One wave per (b,h).
// Q/K/V 32x64 bf16 tiles DMA'd by TDM (row = 32 DW -> interval 4, pad 2 DW).
// ---------------------------------------------------------------------------
__global__ __launch_bounds__(32)
void linattn_scan_kernel(const __bf16* __restrict__ Q,
                         const __bf16* __restrict__ K,
                         const __bf16* __restrict__ V,
                         __bf16* __restrict__ Y) {
  constexpr int LQ = 68;   // bf16 stride: 64 + 4 (TDM pad)
  constexpr int LS = 68;
  constexpr int LP = 36;
  __shared__ __bf16 sQ[32 * LQ], sK[32 * LQ], sV[32 * LQ];
  __shared__ __bf16 sS[64 * LS];
  __shared__ float  sP[32 * LP];
  __shared__ float  sDen[32];
  __shared__ float  sZ[64];

  const int bh = blockIdx.x;               // 0..63
  const int b  = bh >> 4, h = bh & 15;
  const size_t base = (size_t)bh * Tc * HDc;
  const __bf16* Qp = Q + base;
  const __bf16* Kp = K + base;
  const __bf16* Vp = V + base;
  const int lane = threadIdx.x;

  v8f Sa[4][4];
#pragma unroll
  for (int mt = 0; mt < 4; ++mt)
#pragma unroll
    for (int nt = 0; nt < 4; ++nt) Sa[mt][nt] = vzero();
  for (int i = lane; i < 64 * LS; i += 32) sS[i] = (__bf16)0.f;
  sZ[lane] = 0.f; sZ[lane + 32] = 0.f;
  __syncthreads();

  const int mhalf = (lane >> 4) * 8;
  const int ncol  = lane & 15;

  for (int blk = 0; blk < Tc / 32; ++blk) {
    const size_t boff = (size_t)blk * 32 * HDc;
    tdm_load_2d((unsigned)(uintptr_t)sQ, Qp + boff, /*2B*/1, 64, 32, 64, 4, 1);
    tdm_load_2d((unsigned)(uintptr_t)sK, Kp + boff, /*2B*/1, 64, 32, 64, 4, 1);
    tdm_load_2d((unsigned)(uintptr_t)sV, Vp + boff, /*2B*/1, 64, 32, 64, 4, 1);
    __builtin_amdgcn_s_wait_tensorcnt(0);
    __syncthreads();

    // ---- P = Q K^T with causal mask (keep s <= t)
#pragma unroll
    for (int mt = 0; mt < 2; ++mt)
#pragma unroll
      for (int nt = 0; nt < 2; ++nt) {
        if (nt > mt) {
#pragma unroll
          for (int i = 0; i < 8; ++i)
            sP[(mt * 16 + i + mhalf) * LP + nt * 16 + ncol] = 0.f;
        } else {
          v8f c = vzero();
#pragma unroll
          for (int kc = 0; kc < 2; ++kc) {
            const v16bf aq = afrag(sQ, LQ, mt * 16 + ncol, kc * 32, lane);
            const v16bf bk = bfragT(sK, LQ, nt * 16 + ncol, kc * 32, lane);
            c = wmma_bf(aq, bk, c);
          }
#pragma unroll
          for (int i = 0; i < 8; ++i) {
            const int tq = mt * 16 + i + mhalf, ts = nt * 16 + ncol;
            sP[tq * LP + ts] = (ts <= tq) ? c[i] : 0.f;
          }
        }
      }
    __syncthreads();

    // ---- denominator: rowsum(P_masked) + q.z_prev + eps
    {
      float den = EPSc;
#pragma unroll 4
      for (int s = 0; s < 32; ++s) den += sP[lane * LP + s];
#pragma unroll 4
      for (int d = 0; d < 64; ++d) den += (float)sQ[lane * LQ + d] * sZ[d];
      sDen[lane] = den;
    }
    __syncthreads();

    // ---- O = P@V + Q@S, divide, store
    v16bf aP[2], aQ2[2][2];
#pragma unroll
    for (int mt = 0; mt < 2; ++mt) {
      aP[mt]     = afragF(sP, LP, mt * 16 + ncol, 0, lane);
      aQ2[mt][0] = afrag(sQ, LQ, mt * 16 + ncol, 0, lane);
      aQ2[mt][1] = afrag(sQ, LQ, mt * 16 + ncol, 32, lane);
    }
#pragma unroll
    for (int nt = 0; nt < 4; ++nt) {
      const int dv = nt * 16 + ncol;
      const v16bf bV  = bfragR(sV, LQ, dv, 0, lane);
      const v16bf bS0 = bfragR(sS, LS, dv, 0, lane);
      const v16bf bS1 = bfragR(sS, LS, dv, 32, lane);
#pragma unroll
      for (int mt = 0; mt < 2; ++mt) {
        v8f c = vzero();
        c = wmma_bf(aP[mt], bV, c);
        c = wmma_bf(aQ2[mt][0], bS0, c);
        c = wmma_bf(aQ2[mt][1], bS1, c);
#pragma unroll
        for (int i = 0; i < 8; ++i) {
          const int tq = mt * 16 + i + mhalf;
          const float o = c[i] / sDen[tq];
          const int t = blk * 32 + tq;
          Y[((size_t)(b * Tc + t)) * Dc + h * HDc + nt * 16 + ncol] = (__bf16)o;
        }
      }
    }
    __syncthreads();   // all reads of sS/sZ done before update

    // ---- state update: S += K^T @ V ; z += colsum(K)
#pragma unroll
    for (int mt = 0; mt < 4; ++mt) {
      const v16bf aK = afragT(sK, LQ, mt * 16 + ncol, 0, lane);
#pragma unroll
      for (int nt = 0; nt < 4; ++nt) {
        const v16bf bV = bfragR(sV, LQ, nt * 16 + ncol, 0, lane);
        Sa[mt][nt] = wmma_bf(aK, bV, Sa[mt][nt]);
      }
    }
    {
      float z0 = 0.f, z1 = 0.f;
#pragma unroll 4
      for (int t = 0; t < 32; ++t) {
        z0 += (float)sK[t * LQ + lane];
        z1 += (float)sK[t * LQ + lane + 32];
      }
      sZ[lane] += z0;
      sZ[lane + 32] += z1;
    }
#pragma unroll
    for (int mt = 0; mt < 4; ++mt)
#pragma unroll
      for (int nt = 0; nt < 4; ++nt)
#pragma unroll
        for (int i = 0; i < 8; ++i)
          sS[(mt * 16 + i + mhalf) * LS + nt * 16 + ncol] = (__bf16)Sa[mt][nt][i];
    __syncthreads();
  }
}

// ---------------------------------------------------------------------------
// Kernel 3: out = Y(bf16) @ Wo(f32 -> bf16), f32 result.
// Y tile (128x32 bf16, row = 16 DW -> interval 3) and Wo tile DMA'd by TDM.
// ---------------------------------------------------------------------------
__global__ __launch_bounds__(256)
void out_proj_kernel(const __bf16* __restrict__ Y,
                     const float* __restrict__ Wo,
                     float* __restrict__ out) {
  constexpr int LDY = 36;   // bf16 stride: 32 + 4 (TDM pad = 2 DW)
  constexpr int LDW = 130;  // f32 stride
  __shared__ __bf16 sY[128 * LDY];
  __shared__ float  sW[32 * LDW];

  const int rowTile = blockIdx.x;
  const int colTile = blockIdx.y;
  const int tid  = threadIdx.x;
  const int lane = tid & 31;
  const int wave = tid >> 5;
  const int wm   = wave >> 2;
  const int wn   = wave & 3;

  v8f acc[4][2];
#pragma unroll
  for (int mt = 0; mt < 4; ++mt)
#pragma unroll
    for (int nt = 0; nt < 2; ++nt) acc[mt][nt] = vzero();

  for (int k0 = 0; k0 < Dc; k0 += 32) {
    if (wave == 0) {
      tdm_load_2d((unsigned)(uintptr_t)sY,
                  Y + (size_t)(rowTile * 128) * Dc + k0,
                  /*2B*/1, /*d0*/32, /*d1*/128, /*stride*/Dc, /*intv*/3, /*pad*/1);
      tdm_load_2d((unsigned)(uintptr_t)sW,
                  Wo + (size_t)k0 * Dc + colTile * 128,
                  /*4B*/2, /*d0*/128, /*d1*/32, /*stride*/Dc, /*intv*/6, /*pad*/1);
      __builtin_amdgcn_s_wait_tensorcnt(0);
    }
    __syncthreads();

    v16bf bfr[2];
#pragma unroll
    for (int nt = 0; nt < 2; ++nt)
      bfr[nt] = bfragRF(sW, LDW, wn * 32 + nt * 16 + (lane & 15), 0, lane);
#pragma unroll
    for (int mt = 0; mt < 4; ++mt) {
      const v16bf afr = afrag(sY, LDY, wm * 64 + mt * 16 + (lane & 15), 0, lane);
#pragma unroll
      for (int nt = 0; nt < 2; ++nt) acc[mt][nt] = wmma_bf(afr, bfr[nt], acc[mt][nt]);
    }
    __syncthreads();
  }

  const int mhalf = (lane >> 4) * 8;
  const int n     = lane & 15;
#pragma unroll
  for (int mt = 0; mt < 4; ++mt)
#pragma unroll
    for (int nt = 0; nt < 2; ++nt)
#pragma unroll
      for (int i = 0; i < 8; ++i) {
        const int grow = rowTile * 128 + wm * 64 + mt * 16 + i + mhalf;
        const int gcol = colTile * 128 + wn * 32 + nt * 16 + n;
        out[(size_t)grow * Dc + gcol] = acc[mt][nt][i];
      }
}

// ---------------------------------------------------------------------------
extern "C" void kernel_launch(void* const* d_in, const int* in_sizes, int n_in,
                              void* d_out, int out_size, void* d_ws, size_t ws_size,
                              hipStream_t stream) {
  const float* x  = (const float*)d_in[0];
  const float* Wq = (const float*)d_in[1];
  const float* Wk = (const float*)d_in[2];
  const float* Wv = (const float*)d_in[3];
  const float* Wo = (const float*)d_in[4];
  float* out = (float*)d_out;

  // workspace (bf16): Q | K | V | Y  — 4 x 16,777,216 elems = 128 MB
  const size_t nQKV = (size_t)Bc * Hc * Tc * HDc;
  __bf16* Q = (__bf16*)d_ws;
  __bf16* K = Q + nQKV;
  __bf16* V = K + nQKV;
  __bf16* Y = V + nQKV;

  dim3 g1(Nc / 128, Dc / 128, 3);
  qkv_proj_kernel<<<g1, 256, 0, stream>>>(x, Wq, Wk, Wv, Q, K, V);

  linattn_scan_kernel<<<Bc * Hc, 32, 0, stream>>>(Q, K, V, Y);

  dim3 g3(Nc / 128, Dc / 128);
  out_proj_kernel<<<g3, 256, 0, stream>>>(Y, Wo, out);
}
